// SoftMoEBatchedDual_26800595927259
// MI455X (gfx1250) — matrix operand order repaired and checked by reading, not compiled, not run on hardware
//
#include <hip/hip_runtime.h>
#include <hip/hip_bf16.h>

// Problem dims (fixed by the reference)
#define BB    8
#define MM    2048
#define DD    1024
#define FF    2048
#define NE    32
#define PP    4
#define SLOTS 128   // NE*PP

#define SA_STRIDE 40   // padded LDS stride for the 32x32 A slab (16B-aligned, fewer bank conflicts)

typedef __attribute__((ext_vector_type(16))) _Float16 v16h;
typedef __attribute__((ext_vector_type(8)))  float    v8f;

// ---------------------------------------------------------------------------
// Two-wave (64-thread) block GEMM: block computes a 32x64 f32 output tile of
// C = A(32xK,f32) * B(Kx64,f32). Each wave owns 16 rows x 64 cols: per K-step
// it builds ONE A fragment plus FOUR distinct B fragments, then issues four
// v_wmma_f32_16x16x32_f16 back-to-back (disjoint operands -> no inter-WMMA
// hazard NOPs), f32 accumulate. Tiles staged through LDS with coalesced
// float4 loads; f32->f16 conversion happens in-register on LDS read.
//
// Fragment layouts per CDNA5 ISA 7.12.2 (wave32):
//   A 16x32 f16 : lane l (row = l&15, hi = l>>4), vgpr v, half h:
//                 K = ((v>>2)<<4) + (hi<<3) + ((v&3)<<1) + h
//   B 32x16 f16 : lane l (col = l&15, hi = l>>4), vgpr v, half h:
//                 K = ((v>>2)<<4) + (hi<<3) + ((v&3)<<1) + h
//   C 16x16 f32 : lane l, vgpr v: row = v + ((l>>4)<<3), col = l&15
// ---------------------------------------------------------------------------
__device__ __forceinline__ void wmma_gemm_32x64(const float* __restrict__ A, int lda,
                                                const float* __restrict__ B, int ldb,
                                                int K, v8f acc[4]) {
  __shared__ float sA[32 * SA_STRIDE];   // 32 rows x 32 K (padded)
  __shared__ float sB[32 * 64];          // 32 K  x 64 cols

  const int tid = threadIdx.x;      // 0..63
  const int l   = tid & 31;
  const int w   = tid >> 5;         // wave id: row half of the block tile
  const int lm  = l & 15;
  const int hi  = l >> 4;

  for (int k0 = 0; k0 < K; k0 += 32) {
    // --- stage A (32x32): 256 float4 over 64 threads ---
#pragma unroll
    for (int t = 0; t < 4; ++t) {
      const int j  = t * 64 + tid;              // 0..255
      const int ar = j >> 3, ac = (j & 7) << 2; // 8 float4 per row
      *(float4*)(sA + ar * SA_STRIDE + ac) =
          *(const float4*)(A + (size_t)ar * lda + k0 + ac);
    }
    // --- stage B (32x64): 512 float4 over 64 threads ---
#pragma unroll
    for (int t = 0; t < 8; ++t) {
      const int j  = t * 64 + tid;               // 0..511
      const int br = j >> 4, bc = (j & 15) << 2; // 16 float4 per row
      *(float4*)(sB + br * 64 + bc) =
          *(const float4*)(B + (size_t)(k0 + br) * ldb + bc);
    }
    // prefetch next K-tile into cache (global_prefetch_b8)
    if (k0 + 32 < K) {
      __builtin_prefetch(A + (size_t)l * lda + k0 + 32, 0, 3);
      __builtin_prefetch(B + (size_t)(k0 + 32 + l) * ldb + w * 64, 0, 3);
    }
    __syncthreads();

    // --- one A fragment per wave ---
    v16h af;
#pragma unroll
    for (int v = 0; v < 8; ++v) {
      const int ka = ((v >> 2) << 4) + (hi << 3) + ((v & 3) << 1);
      af[2 * v]     = (_Float16)sA[(w * 16 + lm) * SA_STRIDE + ka];
      af[2 * v + 1] = (_Float16)sA[(w * 16 + lm) * SA_STRIDE + ka + 1];
    }
    // --- four B fragments in DISTINCT registers (kills WMMA->VALU WAR nops) ---
    v16h bf[4];
#pragma unroll
    for (int u = 0; u < 4; ++u) {
#pragma unroll
      for (int v = 0; v < 8; ++v) {
        const int kb = ((v >> 2) << 4) + (hi << 3) + ((v & 3) << 1);
        bf[u][2 * v]     = (_Float16)sB[kb * 64 + u * 16 + lm];
        bf[u][2 * v + 1] = (_Float16)sB[(kb + 1) * 64 + u * 16 + lm];
      }
    }
    // --- four WMMAs back-to-back; disjoint operands, no hazard stalls ---
#pragma unroll
    for (int u = 0; u < 4; ++u) {
      acc[u] = __builtin_amdgcn_wmma_f32_16x16x32_f16(
          /*neg_a=*/false, af, /*neg_b=*/false, bf[u],
          /*c_mod=*/(short)0, acc[u], /*reuse_a=*/false, /*reuse_b=*/false);
    }
    __syncthreads();
  }
}

// ---------------------------------------------------------------------------
// K1: logits[b,m,s] = sum_d x[b,m,d] * phi[d,s]   (phi is (D,128) row-major)
// grid = B * (M/32) * (128/64) = 1024, block = 64
// ---------------------------------------------------------------------------
__global__ void k_logits(const float* __restrict__ x, const float* __restrict__ phi,
                         float* __restrict__ logits) {
  const int blk  = blockIdx.x;
  const int b    = blk >> 7;        // 128 block-tiles per batch
  const int r    = blk & 127;
  const int rowT = r >> 1;          // 0..63
  const int colT = r & 1;

  const float* A  = x + (size_t)b * MM * DD + (size_t)rowT * 32 * DD;
  const float* Bm = phi + colT * 64;
  v8f acc[4] = {};
  wmma_gemm_32x64(A, DD, Bm, SLOTS, DD, acc);

  const int l = threadIdx.x & 31, w = threadIdx.x >> 5;
  const int lm = l & 15, rb = (l >> 4) << 3;
  float* C = logits + (size_t)b * MM * SLOTS + (size_t)rowT * 32 * SLOTS + colT * 64;
#pragma unroll
  for (int u = 0; u < 4; ++u)
#pragma unroll
    for (int v = 0; v < 8; ++v)
      C[(size_t)(w * 16 + rb + v) * SLOTS + u * 16 + lm] = acc[u][v];
}

// ---------------------------------------------------------------------------
// K2: dispatch softmax over m (axis=1). Writes TRANSPOSED: dispT[b,s,m].
// grid = B*128 blocks, 256 threads; each block owns one (b,s) column.
// ---------------------------------------------------------------------------
__global__ void k_dispatch_softmax(const float* __restrict__ logits,
                                   float* __restrict__ dispT) {
  const int b = blockIdx.x >> 7;
  const int s = blockIdx.x & 127;
  const float* base = logits + (size_t)b * MM * SLOTS + s;
  const int t = threadIdx.x;

  float vals[8];
  float mx = -INFINITY;
#pragma unroll
  for (int i = 0; i < 8; ++i) {
    vals[i] = base[(size_t)(t + i * 256) * SLOTS];
    mx = fmaxf(mx, vals[i]);
  }
  __shared__ float red[256];
  red[t] = mx;
  __syncthreads();
  for (int off = 128; off > 0; off >>= 1) {
    if (t < off) red[t] = fmaxf(red[t], red[t + off]);
    __syncthreads();
  }
  mx = red[0];
  __syncthreads();

  float sum = 0.f;
#pragma unroll
  for (int i = 0; i < 8; ++i) {
    vals[i] = __expf(vals[i] - mx);
    sum += vals[i];
  }
  red[t] = sum;
  __syncthreads();
  for (int off = 128; off > 0; off >>= 1) {
    if (t < off) red[t] += red[t + off];
    __syncthreads();
  }
  const float inv = 1.0f / red[0];

  float* outp = dispT + ((size_t)b * SLOTS + s) * MM;
#pragma unroll
  for (int i = 0; i < 8; ++i) outp[t + i * 256] = vals[i] * inv;
}

// ---------------------------------------------------------------------------
// K3: combine softmax over the 128 slots (row softmax). One wave per row.
// grid = B*M/8 blocks of 256 (8 waves/block).
// ---------------------------------------------------------------------------
__global__ void k_combine_softmax(const float* __restrict__ logits,
                                  float* __restrict__ comb) {
  const int row  = blockIdx.x * 8 + (threadIdx.x >> 5);
  const int lane = threadIdx.x & 31;
  const float4 v = *(const float4*)(logits + (size_t)row * SLOTS + lane * 4);

  float mx = fmaxf(fmaxf(v.x, v.y), fmaxf(v.z, v.w));
#pragma unroll
  for (int off = 16; off > 0; off >>= 1) mx = fmaxf(mx, __shfl_xor(mx, off, 32));

  float e0 = __expf(v.x - mx), e1 = __expf(v.y - mx);
  float e2 = __expf(v.z - mx), e3 = __expf(v.w - mx);
  float sum = e0 + e1 + e2 + e3;
#pragma unroll
  for (int off = 16; off > 0; off >>= 1) sum += __shfl_xor(sum, off, 32);

  const float inv = 1.0f / sum;
  float4 o; o.x = e0 * inv; o.y = e1 * inv; o.z = e2 * inv; o.w = e3 * inv;
  *(float4*)(comb + (size_t)row * SLOTS + lane * 4) = o;
}

// ---------------------------------------------------------------------------
// K4: xs = dispatchT(128x2048) * X(2048x1024), per batch. Output re-laid as
// xs2[n, b*4+p, d] so each expert's M=32 GEMM is a contiguous slab.
// grid = B * (128/32) * (1024/64) = 512, block = 64
// ---------------------------------------------------------------------------
__global__ void k_xs(const float* __restrict__ dispT, const float* __restrict__ x,
                     float* __restrict__ xs2) {
  const int blk  = blockIdx.x;
  const int b    = blk >> 6;       // 64 block-tiles per batch
  const int r    = blk & 63;
  const int rowT = r >> 4;         // 0..3
  const int colT = r & 15;

  const float* A  = dispT + (size_t)b * SLOTS * MM + (size_t)rowT * 32 * MM;
  const float* Bm = x + (size_t)b * MM * DD + colT * 64;
  v8f acc[4] = {};
  wmma_gemm_32x64(A, MM, Bm, DD, MM, acc);

  const int l = threadIdx.x & 31, w = threadIdx.x >> 5;
  const int lm = l & 15, rb = (l >> 4) << 3;
#pragma unroll
  for (int u = 0; u < 4; ++u)
#pragma unroll
    for (int v = 0; v < 8; ++v) {
      const int s   = rowT * 32 + w * 16 + rb + v;   // slot 0..127
      const int n   = s >> 2, p = s & 3;
      const int col = colT * 64 + u * 16 + lm;
      xs2[(size_t)(n * 32 + b * 4 + p) * DD + col] = acc[u][v];
    }
}

// ---------------------------------------------------------------------------
// K5: h = relu(xs2[n](32x1024) * w1[n](1024x2048) + b1[n])^2
// grid = NE * (2048/64) = 1024, block = 64
// ---------------------------------------------------------------------------
__global__ void k_h(const float* __restrict__ xs2, const float* __restrict__ w1,
                    const float* __restrict__ b1, float* __restrict__ h) {
  const int blk  = blockIdx.x;
  const int n    = blk >> 5;       // 32 column tiles per expert
  const int colT = blk & 31;

  const float* A  = xs2 + (size_t)n * 32 * DD;
  const float* Bm = w1 + (size_t)n * DD * FF + colT * 64;
  v8f acc[4] = {};
  wmma_gemm_32x64(A, DD, Bm, FF, DD, acc);

  const int l = threadIdx.x & 31, w = threadIdx.x >> 5;
  const int lm = l & 15, rb = (l >> 4) << 3;
  const float* bias = b1 + (size_t)n * FF;
#pragma unroll
  for (int u = 0; u < 4; ++u)
#pragma unroll
    for (int v = 0; v < 8; ++v) {
      const int row = w * 16 + rb + v;              // 0..31 (= b*4+p)
      const int col = colT * 64 + u * 16 + lm;
      float val = acc[u][v] + bias[col];
      val = val > 0.f ? val * val : 0.f;            // relu^2
      h[(size_t)(n * 32 + row) * FF + col] = val;
    }
}

// ---------------------------------------------------------------------------
// K6: y = h[n](32x2048) * w2[n](2048x1024) + b2[n]; output laid as y[b,s,d].
// grid = NE * (1024/64) = 512, block = 64
// ---------------------------------------------------------------------------
__global__ void k_y(const float* __restrict__ h, const float* __restrict__ w2,
                    const float* __restrict__ b2, float* __restrict__ y) {
  const int blk  = blockIdx.x;
  const int n    = blk >> 4;       // 16 column tiles per expert
  const int colT = blk & 15;

  const float* A  = h + (size_t)n * 32 * FF;
  const float* Bm = w2 + (size_t)n * FF * DD + colT * 64;
  v8f acc[4] = {};
  wmma_gemm_32x64(A, FF, Bm, DD, FF, acc);

  const int l = threadIdx.x & 31, w = threadIdx.x >> 5;
  const int lm = l & 15, rb = (l >> 4) << 3;
  const float* bias = b2 + (size_t)n * DD;
#pragma unroll
  for (int u = 0; u < 4; ++u)
#pragma unroll
    for (int v = 0; v < 8; ++v) {
      const int row = w * 16 + rb + v;              // 0..31 = b*4 + p
      const int b = row >> 2, p = row & 3;
      const int col = colT * 64 + u * 16 + lm;
      y[(size_t)(b * SLOTS + n * 4 + p) * DD + col] = acc[u][v] + bias[col];
    }
}

// ---------------------------------------------------------------------------
// K7: out = combine(2048x128) * y[b](128x1024), per batch.
// grid = B * (2048/32) * (1024/64) = 8192, block = 64
// ---------------------------------------------------------------------------
__global__ void k_out(const float* __restrict__ comb, const float* __restrict__ y,
                      float* __restrict__ out) {
  const int blk  = blockIdx.x;
  const int b    = blk >> 10;      // 1024 block-tiles per batch
  const int r    = blk & 1023;
  const int rowT = r >> 4;         // 0..63
  const int colT = r & 15;

  const float* A  = comb + (size_t)b * MM * SLOTS + (size_t)rowT * 32 * SLOTS;
  const float* Bm = y + (size_t)b * SLOTS * DD + colT * 64;
  v8f acc[4] = {};
  wmma_gemm_32x64(A, SLOTS, Bm, DD, SLOTS, acc);

  const int l = threadIdx.x & 31, w = threadIdx.x >> 5;
  const int lm = l & 15, rb = (l >> 4) << 3;
  float* C = out + (size_t)b * MM * DD + (size_t)rowT * 32 * DD + colT * 64;
#pragma unroll
  for (int u = 0; u < 4; ++u)
#pragma unroll
    for (int v = 0; v < 8; ++v)
      C[(size_t)(w * 16 + rb + v) * DD + u * 16 + lm] = acc[u][v];
}

// ---------------------------------------------------------------------------
extern "C" void kernel_launch(void* const* d_in, const int* in_sizes, int n_in,
                              void* d_out, int out_size, void* d_ws, size_t ws_size,
                              hipStream_t stream) {
  const float* x   = (const float*)d_in[0];
  const float* phi = (const float*)d_in[1];
  const float* w1  = (const float*)d_in[2];
  const float* b1  = (const float*)d_in[3];
  const float* w2  = (const float*)d_in[4];
  const float* b2  = (const float*)d_in[5];
  float* out = (float*)d_out;

  // workspace layout (floats): all regions fully rewritten each call
  float* logits = (float*)d_ws;                                   // B*M*128  = 8 MB
  float* dispT  = logits + (size_t)BB * MM * SLOTS;               // B*128*M  = 8 MB
  float* comb   = dispT  + (size_t)BB * SLOTS * MM;               // B*M*128  = 8 MB
  float* xs2    = comb   + (size_t)BB * MM * SLOTS;               // NE*32*D  = 4 MB
  float* hbuf   = xs2    + (size_t)NE * 32 * DD;                  // NE*32*F  = 8 MB
  float* ybuf   = hbuf   + (size_t)NE * 32 * FF;                  // B*128*D  = 4 MB

  k_logits<<<1024, 64, 0, stream>>>(x, phi, logits);
  k_dispatch_softmax<<<BB * SLOTS, 256, 0, stream>>>(logits, dispT);
  k_combine_softmax<<<(BB * MM) / 8, 256, 0, stream>>>(logits, comb);
  k_xs<<<512, 64, 0, stream>>>(dispT, x, xs2);
  k_h<<<1024, 64, 0, stream>>>(xs2, w1, b1, hbuf);
  k_y<<<512, 64, 0, stream>>>(hbuf, w2, b2, ybuf);
  k_out<<<8192, 64, 0, stream>>>(comb, ybuf, out);
}
